// HeteroGATLayer_54443005444388
// MI455X (gfx1250) — matrix operand order repaired and checked by reading, not compiled, not run on hardware
//
#include <hip/hip_runtime.h>
#include <hip/hip_bf16.h>

#define F_IN 128
#define HD   256   // H * D
#define H    4
#define D    64
#define T_TYPES 3

typedef __attribute__((ext_vector_type(16))) __bf16 v16bf;
typedef __attribute__((ext_vector_type(8)))  __bf16 v8bf;
typedef __attribute__((ext_vector_type(8)))  float  v8f;

__device__ __forceinline__ __bf16 f32_to_bf16(float f) {
  unsigned u = __float_as_uint(f);
  unsigned r = u + 0x7FFFu + ((u >> 16) & 1u);   // round-to-nearest-even
  unsigned short hs = (unsigned short)(r >> 16);
  __bf16 b;
  __builtin_memcpy(&b, &hs, 2);
  return b;
}

// Monotone float ordering encode/decode for atomicMax on unsigned
__device__ __forceinline__ unsigned f_enc(float f) {
  unsigned u = __float_as_uint(f);
  return (u & 0x80000000u) ? ~u : (u | 0x80000000u);
}
__device__ __forceinline__ float f_dec(unsigned u) {
  unsigned b = (u & 0x80000000u) ? (u ^ 0x80000000u) : ~u;
  return __uint_as_float(b);
}

__device__ __forceinline__ float leaky(float v) {
  return v >= 0.f ? v : 0.2f * v;
}

// ---------------- one-time packs -------------------------------------------
// xp[n][k] = bf16(x[n][k])
__global__ void pack_x_bf16(const float* __restrict__ x,
                            __bf16* __restrict__ xp, int total) {
  int idx = blockIdx.x * blockDim.x + threadIdx.x;
  if (idx >= total) return;
  xp[idx] = f32_to_bf16(x[idx]);
}

// wT[t][n][k] = bf16(W[t][k][n])   (transpose so B-fragments load contiguously)
__global__ void pack_wT_bf16(const float* __restrict__ W,
                             __bf16* __restrict__ wT, int total) {
  int idx = blockIdx.x * blockDim.x + threadIdx.x;
  if (idx >= total) return;
  int k = idx & (F_IN - 1);
  int n = (idx >> 7) & (HD - 1);
  int t = idx >> 15;
  wT[idx] = f32_to_bf16(W[((size_t)t * F_IN + k) * HD + n]);
}

// ---------------- GEMM: feat[N,256] = x[N,128] @ W[128,256] via WMMA bf16 ----
// One wave computes one 16x16 tile; K=128 -> 4 wmma steps; all loads are b128.
__global__ void gemm_x_w_wmma(const __bf16* __restrict__ xp,   // [N,128]
                              const __bf16* __restrict__ wT,   // [256,128] (this type)
                              float* __restrict__ feat,
                              int n_tiles_total) {
  int wave = (blockIdx.x * blockDim.x + threadIdx.x) >> 5;
  if (wave >= n_tiles_total) return;              // whole wave uniform
  int lane = threadIdx.x & 31;
  int row_tile = wave >> 4;                       // 256/16 = 16 column tiles
  int col_tile = wave & 15;
  int m_base = row_tile * 16;
  int n_base = col_tile * 16;
  int half   = lane >> 4;                         // 0: lanes 0-15, 1: lanes 16-31
  int lo     = lane & 15;

  const __bf16* arow = xp + (size_t)(m_base + lo) * F_IN;
  const __bf16* bcol = wT + (size_t)(n_base + lo) * F_IN;

  v8f acc = {};
  #pragma unroll
  for (int ks = 0; ks < F_IN / 32; ++ks) {
    int k0 = ks * 32;
    // A (16x32): lane lo holds row m; elems 0-7 -> K=k0+half*8+i,
    //            elems 8-15 -> K=k0+16+half*8+(i-8)  => two contiguous 16B chunks
    v8bf a0 = *(const v8bf*)(arow + k0 + half * 8);
    v8bf a1 = *(const v8bf*)(arow + k0 + 16 + half * 8);
    v16bf a = __builtin_shufflevector(a0, a1, 0, 1, 2, 3, 4, 5, 6, 7,
                                      8, 9, 10, 11, 12, 13, 14, 15);
    // B (32x16): lane lo holds col n; elems 0-7 -> K=k0+half*8+i, elems 8-15 -> K=k0+16+half*8+(i-8)
    v8bf b0 = *(const v8bf*)(bcol + k0 + half * 8);
    v8bf b1 = *(const v8bf*)(bcol + k0 + 16 + half * 8);
    v16bf b = __builtin_shufflevector(b0, b1, 0, 1, 2, 3, 4, 5, 6, 7,
                                      8, 9, 10, 11, 12, 13, 14, 15);
    acc = __builtin_amdgcn_wmma_f32_16x16x32_bf16(false, a, false, b,
                                                  (short)0, acc, false, false);
  }
  // C/D layout: element r -> m = r + half*8, n = lo
  #pragma unroll
  for (int r = 0; r < 8; ++r) {
    int m = m_base + r + half * 8;
    feat[(size_t)m * HD + n_base + lo] = acc[r];
  }
}

// ---------------- el/er projections: [N,H] = sum_d feat[n,h,d]*attn[h,d] -----
__global__ void compute_elr(const float* __restrict__ feat,
                            const float* __restrict__ al,
                            const float* __restrict__ ar,
                            float* __restrict__ el, float* __restrict__ er,
                            int NH) {
  int idx = blockIdx.x * blockDim.x + threadIdx.x;
  if (idx >= NH) return;
  int n = idx >> 2, h = idx & 3;
  const float* f = feat + (size_t)n * HD + h * D;
  const float* a = al + h * D;
  const float* b = ar + h * D;
  float sl = 0.f, sr = 0.f;
  #pragma unroll 8
  for (int d = 0; d < D; ++d) { float v = f[d]; sl += v * a[d]; sr += v * b[d]; }
  el[idx] = sl;
  er[idx] = sr;
}

// ---------------- per-type init: m = enc(-inf), denom = 0 --------------------
__global__ void init_md(unsigned* __restrict__ m_enc, float* __restrict__ denom,
                        int NH) {
  int idx = blockIdx.x * blockDim.x + threadIdx.x;
  if (idx >= NH) return;
  m_enc[idx] = 0x007FFFFFu;   // f_enc(-inf)
  denom[idx] = 0.f;
}

// ---------------- segment max over dst ---------------------------------------
__global__ void edge_max(const int* __restrict__ src, const int* __restrict__ dst,
                         const float* __restrict__ el, const float* __restrict__ er,
                         unsigned* __restrict__ m_enc, int EH) {
  int idx = blockIdx.x * blockDim.x + threadIdx.x;
  if (idx >= EH) return;
  int e = idx >> 2, h = idx & 3;
  int s = src[e], d = dst[e];
  float v = leaky(el[s * H + h] + er[d * H + h]);
  atomicMax(&m_enc[d * H + h], f_enc(v));
}

// decode max in place (uint -> float view), non-finite -> 0 (zero-indegree)
__global__ void finalize_m(unsigned* __restrict__ m_buf, int NH) {
  int idx = blockIdx.x * blockDim.x + threadIdx.x;
  if (idx >= NH) return;
  float v = f_dec(m_buf[idx]);
  unsigned b = __float_as_uint(v);
  if ((b & 0x7F800000u) == 0x7F800000u) v = 0.f;   // inf / nan guard
  ((float*)m_buf)[idx] = v;
}

// ---------------- softmax denominator ----------------------------------------
__global__ void edge_denom(const int* __restrict__ src, const int* __restrict__ dst,
                           const float* __restrict__ el, const float* __restrict__ er,
                           const float* __restrict__ m, float* __restrict__ denom,
                           int EH) {
  int idx = blockIdx.x * blockDim.x + threadIdx.x;
  if (idx >= EH) return;
  int e = idx >> 2, h = idx & 3;
  int s = src[e], d = dst[e];
  float v = leaky(el[s * H + h] + er[d * H + h]);
  atomicAdd(&denom[d * H + h], __expf(v - m[d * H + h]));
}

// ---------------- message scatter: one wave per edge -------------------------
// lane -> head h = lane>>3 ; 8 contiguous feature floats d0 = (lane&7)*8
__global__ void edge_scatter(const int* __restrict__ src, const int* __restrict__ dst,
                             const float* __restrict__ el, const float* __restrict__ er,
                             const float* __restrict__ m, const float* __restrict__ denom,
                             const float* __restrict__ feat, float* __restrict__ rst,
                             int E) {
  int wave = (blockIdx.x * blockDim.x + threadIdx.x) >> 5;
  if (wave >= E) return;
  int lane = threadIdx.x & 31;
  int s = src[wave], d = dst[wave];
  int h = lane >> 3;
  float v  = leaky(el[s * H + h] + er[d * H + h]);
  float dn = denom[d * H + h];
  float alpha = __expf(v - m[d * H + h]) / (dn > 0.f ? dn : 1.f);
  int off = h * D + (lane & 7) * 8;
  const float* fs = feat + (size_t)s * HD + off;
  float*       ro = rst  + (size_t)d * HD + off;
  float4 f0 = *(const float4*)(fs);
  float4 f1 = *(const float4*)(fs + 4);
  atomicAdd(ro + 0, alpha * f0.x);
  atomicAdd(ro + 1, alpha * f0.y);
  atomicAdd(ro + 2, alpha * f0.z);
  atomicAdd(ro + 3, alpha * f0.w);
  atomicAdd(ro + 4, alpha * f1.x);
  atomicAdd(ro + 5, alpha * f1.y);
  atomicAdd(ro + 6, alpha * f1.z);
  atomicAdd(ro + 7, alpha * f1.w);
}

// ---------------- output: mean over heads + summed bias ----------------------
__global__ void final_out(const float* __restrict__ rst,
                          const float* __restrict__ bias,
                          float* __restrict__ out, int ND) {
  int idx = blockIdx.x * blockDim.x + threadIdx.x;
  if (idx >= ND) return;
  int n = idx >> 6, d = idx & 63;
  float acc = 0.f;
  #pragma unroll
  for (int h = 0; h < H; ++h) {
    float bsum = 0.f;
    #pragma unroll
    for (int t = 0; t < T_TYPES; ++t)
      bsum += bias[((size_t)t * H + h) * D + d];
    acc += rst[(size_t)n * HD + h * D + d] + bsum;
  }
  out[idx] = acc * (1.0f / H);
}

extern "C" void kernel_launch(void* const* d_in, const int* in_sizes, int n_in,
                              void* d_out, int out_size, void* d_ws, size_t ws_size,
                              hipStream_t stream) {
  const float* x    = (const float*)d_in[0];
  const int*   src  = (const int*)d_in[1];
  const int*   dst  = (const int*)d_in[2];
  const float* W    = (const float*)d_in[3];
  const float* al   = (const float*)d_in[4];
  const float* ar   = (const float*)d_in[5];
  const float* bias = (const float*)d_in[6];

  const int N = in_sizes[0] / F_IN;
  const int E = in_sizes[1] / T_TYPES;
  const int NH = N * H;
  const int EH = E * H;

  // workspace carve-up
  float*    feat  = (float*)d_ws;                       // N*256 f32
  float*    rst   = feat + (size_t)N * HD;              // N*256 f32 (accum over types)
  float*    el    = rst  + (size_t)N * HD;              // N*4
  float*    er    = el   + (size_t)NH;                  // N*4
  unsigned* m     = (unsigned*)(er + (size_t)NH);       // N*4 (uint then float view)
  float*    denom = (float*)(m + (size_t)NH);           // N*4
  __bf16*   xp    = (__bf16*)(denom + (size_t)NH);      // N*128 bf16
  __bf16*   wT    = xp + (size_t)N * F_IN;              // T*256*128 bf16

  hipMemsetAsync(rst, 0, sizeof(float) * (size_t)N * HD, stream);

  dim3 blk(256);
  const int xp_total = N * F_IN;
  const int wt_total = T_TYPES * HD * F_IN;
  pack_x_bf16 <<<(xp_total + 255) / 256, blk, 0, stream>>>(x, xp, xp_total);
  pack_wT_bf16<<<(wt_total + 255) / 256, blk, 0, stream>>>(W, wT, wt_total);

  const int n_tiles = ((N + 15) / 16) * (HD / 16);      // 16x16 output tiles

  for (int t = 0; t < T_TYPES; ++t) {
    const int* src_t = src + (size_t)t * E;
    const int* dst_t = dst + (size_t)t * E;
    const __bf16* wT_t = wT + (size_t)t * HD * F_IN;

    gemm_x_w_wmma<<<(n_tiles + 7) / 8, blk, 0, stream>>>(xp, wT_t, feat, n_tiles);
    compute_elr  <<<(NH + 255) / 256, blk, 0, stream>>>(feat, al + t * H * D,
                                                        ar + t * H * D, el, er, NH);
    init_md      <<<(NH + 255) / 256, blk, 0, stream>>>(m, denom, NH);
    edge_max     <<<(EH + 255) / 256, blk, 0, stream>>>(src_t, dst_t, el, er, m, EH);
    finalize_m   <<<(NH + 255) / 256, blk, 0, stream>>>(m, NH);
    edge_denom   <<<(EH + 255) / 256, blk, 0, stream>>>(src_t, dst_t, el, er,
                                                        (const float*)m, denom, EH);
    edge_scatter <<<(E + 7) / 8, blk, 0, stream>>>(src_t, dst_t, el, er,
                                                   (const float*)m, denom,
                                                   feat, rst, E);
  }

  final_out<<<(N * D + 255) / 256, blk, 0, stream>>>(rst, bias, (float*)d_out, N * D);
}
